// DenoisingNetwork_1322849927184
// MI455X (gfx1250) — matrix operand order repaired and checked by reading, not compiled, not run on hardware
//
#include <hip/hip_runtime.h>
#include <hip/hip_bf16.h>
#include <stdint.h>

// ---- problem constants (match reference) ----
#define NN   10000
#define EE   160000
#define HH   256
#define KKm  20
#define NLAY 5
#define NODEF 64
#define NNTT 10
#define NETT 5

typedef _Float16 h16;
typedef __attribute__((ext_vector_type(8)))  _Float16 h8v;
typedef __attribute__((ext_vector_type(16))) _Float16 h16v;
typedef __attribute__((ext_vector_type(8)))  float    f8v;
typedef __attribute__((ext_vector_type(4)))  unsigned int u32x4;
typedef __attribute__((ext_vector_type(8)))  int          i32x8;
typedef __attribute__((ext_vector_type(4)))  int          i32x4;

// ---------------------------------------------------------------------------
// gfx1250 async data-movement helpers
// ---------------------------------------------------------------------------
// Low 32 bits of a generic shared-memory address == LDS byte offset.
__device__ inline unsigned lds_off_u32(const void* p) {
    return (unsigned)(uintptr_t)p;
}

__device__ inline void async_g2l_b128(unsigned lds_off, const h16* g) {
    asm volatile("global_load_async_to_lds_b128 %0, %1, off"
                 :: "v"(lds_off), "v"(g) : "memory");
}

__device__ inline void wait_async0() {
#if __has_builtin(__builtin_amdgcn_s_wait_asynccnt)
    __builtin_amdgcn_s_wait_asynccnt(0);
#else
    asm volatile("s_wait_asynccnt 0x0" ::: "memory");
#endif
}

// ---------------------------------------------------------------------------
// Elementwise helpers
// ---------------------------------------------------------------------------
__global__ void k_f32_to_f16(const float* __restrict__ in, h16* __restrict__ out, long n) {
    long t = (long)blockIdx.x * 256 + threadIdx.x;
    if (t < n) out[t] = (h16)in[t];
}

__global__ void k_zero_f32(float* __restrict__ p, long n) {
    long t = (long)blockIdx.x * 256 + threadIdx.x;
    if (t < n) p[t] = 0.0f;
}

__global__ void k_transpose_f32_to_f16(const float* __restrict__ in, h16* __restrict__ out,
                                       int rows, int cols) {
    long t = (long)blockIdx.x * 256 + threadIdx.x;
    if (t >= (long)rows * cols) return;
    int r = (int)(t / cols), c = (int)(t % cols);
    out[(size_t)c * rows + r] = (h16)in[t];
}

// h_e[e][h] = edge_attr[e] * eeW[h] + eeb[h]  (stored f16)
__global__ void k_edge_embed(const float* __restrict__ ea, const float* __restrict__ eeW,
                             const float* __restrict__ eeb, h16* __restrict__ he, int ne) {
    long t = (long)blockIdx.x * 256 + threadIdx.x;
    if (t >= (long)ne * HH) return;
    int e = (int)(t >> 8), h = (int)(t & 255);
    he[t] = (h16)(ea[e] * eeW[h] + eeb[h]);
}

// cat[e] = [h_v[dst], h_v[src], h_e[e]]   (f16, 8-wide vector copies)
__global__ void k_build_cat(const h16* __restrict__ hv16, const h16* __restrict__ he16,
                            const int* __restrict__ ei, h16* __restrict__ cat, int ne) {
    long t = (long)blockIdx.x * 256 + threadIdx.x;
    if (t >= (long)ne * 96) return;
    int e = (int)(t / 96);
    int col = (int)(t % 96) * 8;
    int src = ei[e];
    int dst = ei[ne + e];
    h8v v;
    if (col < HH)            v = *(const h8v*)(hv16 + (size_t)dst * HH + col);
    else if (col < 2 * HH)   v = *(const h8v*)(hv16 + (size_t)src * HH + (col - HH));
    else                     v = *(const h8v*)(he16 + (size_t)e * HH + (col - 2 * HH));
    *(h8v*)(cat + (size_t)e * (3 * HH) + col) = v;
}

// agg[dst] += m*a   (segment sum via f32 atomics)
__global__ void k_mul_agg(const h16* __restrict__ m16, const h16* __restrict__ a16,
                          const int* __restrict__ ei, float* __restrict__ agg, int ne) {
    long t = (long)blockIdx.x * 256 + threadIdx.x;
    if (t >= (long)ne * HH) return;
    int e = (int)(t >> 8), h = (int)(t & 255);
    int dst = ei[ne + e];
    atomicAdd(agg + (size_t)dst * HH + h, (float)m16[t] * (float)a16[t]);
}

__global__ void k_build_xcat(const float* __restrict__ hv, const float* __restrict__ agg,
                             h16* __restrict__ xcat, int nn) {
    long t = (long)blockIdx.x * 256 + threadIdx.x;
    if (t >= (long)nn * HH) return;
    int n = (int)(t >> 8), h = (int)(t & 255);
    xcat[(size_t)n * (2 * HH) + h]      = (h16)hv[t];
    xcat[(size_t)n * (2 * HH) + HH + h] = (h16)agg[t];
}

__global__ void k_build_npin(const float* __restrict__ ge, const float* __restrict__ hv,
                             h16* __restrict__ np_in, int nn) {
    long t = (long)blockIdx.x * 256 + threadIdx.x;
    if (t >= (long)nn * HH) return;
    int n = (int)(t >> 8), h = (int)(t & 255);
    np_in[(size_t)n * (2 * HH) + h]      = (h16)ge[h];
    np_in[(size_t)n * (2 * HH) + HH + h] = (h16)hv[t];
}

__global__ void k_build_alin(const float* __restrict__ ge, const float* __restrict__ hv,
                             const int* __restrict__ vt, h16* __restrict__ al_in, int nn) {
    long t = (long)blockIdx.x * 256 + threadIdx.x;
    if (t >= (long)nn * HH) return;
    int n = (int)(t >> 8), h = (int)(t & 255);
    int v = vt[0];
    al_in[(size_t)n * (3 * HH) + h]          = (h16)ge[h];
    al_in[(size_t)n * (3 * HH) + HH + h]     = (h16)hv[(size_t)v * HH + h];
    al_in[(size_t)n * (3 * HH) + 2 * HH + h] = (h16)hv[t];
}

// ---------------------------------------------------------------------------
// WMMA GEMM: C[M,N] = A[M,K](f16) @ B[K,N](f16) + bias[N], optional ReLU.
// Block tile 128x128, 256 threads (8 waves), wave tile 32x64 (2x4 WMMA tiles).
// A tile staged via TDM (tensor_load_to_lds) or async global->LDS; B tile
// staged transposed manually (TDM cannot transpose).
// ---------------------------------------------------------------------------
__global__ __launch_bounds__(256) void k_wmma_gemm(
    const h16* __restrict__ A, const h16* __restrict__ B, const float* __restrict__ bias,
    h16* __restrict__ out16, float* __restrict__ out32, int M, int N, int K, int relu) {
    __shared__ __align__(16) h16 sA[128 * 40];   // row-major, pitch 40 f16 (16+4 dwords)
    __shared__ __align__(16) h16 sBt[128 * 40];  // transposed: [col][k], pitch 40

    const int tid  = threadIdx.x;
    const int lane = tid & 31;
    const int wave = tid >> 5;
    const int row0 = blockIdx.y * 128;
    const int col0 = blockIdx.x * 128;
    const int wm   = (wave & 3) * 32;   // 4 waves along M
    const int wn   = (wave >> 2) * 64;  // 2 waves along N

    f8v acc[2][4] = {};

    for (int kc = 0; kc < K; kc += 32) {
        // ---- stage A: 32x128 (K x M) tile of A[row0.., kc..] into sA ----
#if __has_builtin(__builtin_amdgcn_tensor_load_to_lds)
        if (tid == 0) {
            // D# descriptor: 2D tile, tile_dim0=32 (K), tile_dim1=128 (M rows),
            // data_size=2B, row stride = K elems, LDS pitch padded 16dw + 4dw.
            unsigned long long ga =
                (unsigned long long)(uintptr_t)(A + (size_t)row0 * K + kc);
            unsigned ldsa = lds_off_u32(sA);
            u32x4 g0 = { 1u,                                   // count=1
                         ldsa,                                  // lds_addr
                         (unsigned)ga,                          // global_addr[31:0]
                         (unsigned)((ga >> 32) & 0x01ffffffu) | (2u << 30) }; // addr hi | type=2
            int td0 = K - kc;        // remaining K extent (OOB -> zero fill)
            int td1 = M - row0;      // remaining M extent (OOB -> zero fill)
            i32x8 g1;
            g1[0] = (1 << 16)        // data_size = 2 bytes
                  | (1 << 20)        // pad_enable
                  | (3 << 22)        // pad_interval: 16 dwords
                  | (3 << 25);       // pad_amount: 4 dwords
            g1[1] = (td0 & 0xffff) << 16;                          // tensor_dim0[15:0]
            g1[2] = ((td0 >> 16) & 0xffff) | ((td1 & 0xffff) << 16); // dim0 hi | dim1 lo
            g1[3] = ((td1 >> 16) & 0xffff) | (32 << 16);           // dim1 hi | tile_dim0=32
            g1[4] = 128;                                           // tile_dim1=128, tile_dim2=0
            g1[5] = K;                                             // tensor_dim0_stride lo
            g1[6] = 0;
            g1[7] = 0;
            i32x4 z4 = { 0, 0, 0, 0 };
            i32x8 z8 = { 0, 0, 0, 0, 0, 0, 0, 0 };
            __builtin_amdgcn_tensor_load_to_lds(g0, g1, z4, z4, z8, 0);
        }
#else
        #pragma unroll
        for (int s = 0; s < 2; ++s) {
            int c  = tid + s * 256;        // 0..511 chunks of 8 f16
            int r  = c >> 2;
            int c8 = (c & 3) << 3;
            int gr = row0 + r;
            if (gr < M) {
                async_g2l_b128(lds_off_u32(sA + r * 40 + c8),
                               A + (size_t)gr * K + kc + c8);
            } else {
                h8v z = {};
                *(h8v*)(sA + r * 40 + c8) = z;
            }
        }
#endif
        // ---- stage B transposed: source tile 32xN ----
        #pragma unroll
        for (int s = 0; s < 2; ++s) {
            int c  = tid + s * 256;
            int kr = c >> 4;               // 0..31
            int c8 = (c & 15) << 3;        // 0..120
            int gc = col0 + c8;
            h8v v = {};
            if (gc < N) v = *(const h8v*)(B + (size_t)(kc + kr) * N + gc);
            #pragma unroll
            for (int j = 0; j < 8; ++j) sBt[(c8 + j) * 40 + kr] = v[j];
        }
        // ---- wait for async/TDM A-tile, then rendezvous ----
#if __has_builtin(__builtin_amdgcn_tensor_load_to_lds)
        if (tid == 0) __builtin_amdgcn_s_wait_tensorcnt((short)0);
#else
        wait_async0();
#endif
        __syncthreads();

        // ---- fragments per ISA VGPR layouts ----
        const int rsel = lane & 15;
        const int koff = (lane >> 4) << 3;  // A: 0 or 8
        const int kb   = (lane >> 4) << 3;  // B: 0 or 8
        h16v af[2], bf[4];
        #pragma unroll
        for (int i = 0; i < 2; ++i) {
            const h16* p = sA + (wm + i * 16 + rsel) * 40;
            h8v lo = *(const h8v*)(p + koff);
            h8v hi = *(const h8v*)(p + 16 + koff);
            af[i] = __builtin_shufflevector(lo, hi, 0, 1, 2, 3, 4, 5, 6, 7,
                                            8, 9, 10, 11, 12, 13, 14, 15);
        }
        #pragma unroll
        for (int j = 0; j < 4; ++j) {
            const h16* p = sBt + (wn + j * 16 + rsel) * 40 + kb;
            h8v lo = *(const h8v*)(p);
            h8v hi = *(const h8v*)(p + 16);
            bf[j] = __builtin_shufflevector(lo, hi, 0, 1, 2, 3, 4, 5, 6, 7,
                                            8, 9, 10, 11, 12, 13, 14, 15);
        }
        #pragma unroll
        for (int i = 0; i < 2; ++i)
            #pragma unroll
            for (int j = 0; j < 4; ++j)
                acc[i][j] = __builtin_amdgcn_wmma_f32_16x16x32_f16(
                    false, af[i], false, bf[j], (short)0, acc[i][j], false, false);
        __syncthreads();
    }

    // ---- epilogue: C layout row = 8*(lane/16)+vgpr, col = lane%16 ----
    const int rsel = lane & 15;
    const int rhi  = (lane >> 4) << 3;
    #pragma unroll
    for (int i = 0; i < 2; ++i) {
        #pragma unroll
        for (int j = 0; j < 4; ++j) {
            int c = col0 + wn + j * 16 + rsel;
            if (c >= N) continue;
            float bv = bias ? bias[c] : 0.0f;
            #pragma unroll
            for (int e = 0; e < 8; ++e) {
                int r = row0 + wm + i * 16 + rhi + e;
                if (r >= M) continue;
                float v = acc[i][j][e] + bv;
                if (relu) v = v > 0.0f ? v : 0.0f;
                if (out16) out16[(size_t)r * N + c] = (h16)v;
                if (out32) out32[(size_t)r * N + c] = v;
            }
        }
    }
}

// ---------------------------------------------------------------------------
// Sequential GRU over the node dimension. One block, 768 threads.
// gh = whh @ h + bhh (768x256 matvec, h in LDS), then gate update on h[0:256].
// ---------------------------------------------------------------------------
__global__ __launch_bounds__(768) void k_gru_scan(
    const float* __restrict__ gi, const float* __restrict__ whh,
    const float* __restrict__ bhh, float* __restrict__ hv, h16* __restrict__ hv16,
    int n_nodes) {
    __shared__ __align__(16) float sh[HH];
    __shared__ float sgh[3 * HH];
    const int t = threadIdx.x;  // 0..767
    if (t < HH) sh[t] = 0.0f;
    __syncthreads();
    const float* wr = whh + (size_t)t * HH;
    const float bb = bhh[t];
    for (int step = 0; step < n_nodes; ++step) {
        float acc = bb;
        #pragma unroll 16
        for (int k4 = 0; k4 < HH / 4; ++k4) {
            float4 w = ((const float4*)wr)[k4];
            float4 h4 = ((const float4*)sh)[k4];
            acc += w.x * h4.x + w.y * h4.y + w.z * h4.z + w.w * h4.w;
        }
        sgh[t] = acc;
        __syncthreads();
        if (t < HH) {
            const float* g = gi + (size_t)step * (3 * HH);
            float r = 1.0f / (1.0f + __expf(-(g[t] + sgh[t])));
            float z = 1.0f / (1.0f + __expf(-(g[HH + t] + sgh[HH + t])));
            float n = tanhf(g[2 * HH + t] + r * sgh[2 * HH + t]);
            float h2 = (1.0f - z) * n + z * sh[t];
            sh[t] = h2;
            hv[(size_t)step * HH + t] = h2;
            hv16[(size_t)step * HH + t] = (h16)h2;
        }
        __syncthreads();
    }
}

// ---------------------------------------------------------------------------
// Heads
// ---------------------------------------------------------------------------
__global__ void k_colmean(const float* __restrict__ hv, float* __restrict__ ge, int nn) {
    __shared__ float red[256];
    int c = blockIdx.x;  // 0..255
    float s = 0.0f;
    for (int r = threadIdx.x; r < nn; r += 256) s += hv[(size_t)r * HH + c];
    red[threadIdx.x] = s;
    __syncthreads();
    for (int o = 128; o > 0; o >>= 1) {
        if (threadIdx.x < o) red[threadIdx.x] += red[threadIdx.x + o];
        __syncthreads();
    }
    if (threadIdx.x == 0) ge[c] = red[0] / (float)nn;
}

// p_v = softmax(nph @ npW2 + npb2), 10 classes
__global__ void k_node_pred(const h16* __restrict__ nph, const float* __restrict__ W2,
                            const float* __restrict__ b2, float* __restrict__ out, int nn) {
    int n = blockIdx.x * 256 + threadIdx.x;
    if (n >= nn) return;
    const h16* x = nph + (size_t)n * HH;
    float lg[NNTT];
    #pragma unroll
    for (int j = 0; j < NNTT; ++j) lg[j] = b2[j];
    for (int h = 0; h < HH; ++h) {
        float xv = (float)x[h];
        #pragma unroll
        for (int j = 0; j < NNTT; ++j) lg[j] += xv * W2[h * NNTT + j];
    }
    float m = lg[0];
    #pragma unroll
    for (int j = 1; j < NNTT; ++j) m = lg[j] > m ? lg[j] : m;
    float s = 0.0f;
    #pragma unroll
    for (int j = 0; j < NNTT; ++j) { lg[j] = __expf(lg[j] - m); s += lg[j]; }
    #pragma unroll
    for (int j = 0; j < NNTT; ++j) out[(size_t)n * NNTT + j] = lg[j] / s;
}

// per-node accumulation of alpha logits (summed over nodes via atomics)
__global__ void k_alpha_acc(const h16* __restrict__ ah, const float* __restrict__ W2,
                            const float* __restrict__ b2, float* __restrict__ acc20, int nn) {
    int n = blockIdx.x * 256 + threadIdx.x;
    if (n >= nn) return;
    const h16* x = ah + (size_t)n * HH;
    float lg[KKm];
    #pragma unroll
    for (int j = 0; j < KKm; ++j) lg[j] = b2[j];
    for (int h = 0; h < HH; ++h) {
        float xv = (float)x[h];
        #pragma unroll
        for (int j = 0; j < KKm; ++j) lg[j] += xv * W2[h * KKm + j];
    }
    #pragma unroll
    for (int j = 0; j < KKm; ++j) atomicAdd(&acc20[j], lg[j]);
}

__global__ void k_alpha_softmax(float* __restrict__ a) {
    if (threadIdx.x == 0) {
        float m = a[0];
        for (int i = 1; i < KKm; ++i) m = a[i] > m ? a[i] : m;
        float s = 0.0f;
        for (int i = 0; i < KKm; ++i) { a[i] = __expf(a[i] - m); s += a[i]; }
        for (int i = 0; i < KKm; ++i) a[i] /= s;
    }
}

// p_e: thread = (node, k); softmax over NET axis, weighted by alphas[k]
__global__ void k_edge_pred(const h16* __restrict__ eh, const float* __restrict__ W2,
                            const float* __restrict__ b2, const float* __restrict__ alphas,
                            float* __restrict__ outpe, int nn) {
    long t = (long)blockIdx.x * 256 + threadIdx.x;
    if (t >= (long)nn * KKm) return;
    int n = (int)(t / KKm), k = (int)(t % KKm);
    const h16* x = eh + (size_t)n * HH;
    float lt[NETT];
    #pragma unroll
    for (int j = 0; j < NETT; ++j) lt[j] = b2[j * KKm + k];
    for (int h = 0; h < HH; ++h) {
        float xv = (float)x[h];
        #pragma unroll
        for (int j = 0; j < NETT; ++j) lt[j] += xv * W2[h * (NETT * KKm) + j * KKm + k];
    }
    float m = lt[0];
    #pragma unroll
    for (int j = 1; j < NETT; ++j) m = lt[j] > m ? lt[j] : m;
    float s = 0.0f;
    #pragma unroll
    for (int j = 0; j < NETT; ++j) { lt[j] = __expf(lt[j] - m); s += lt[j]; }
    float al = alphas[k];
    #pragma unroll
    for (int j = 0; j < NETT; ++j)
        atomicAdd(&outpe[(size_t)n * NETT + j], al * lt[j] / s);
}

// ---------------------------------------------------------------------------
// Host orchestration
// ---------------------------------------------------------------------------
extern "C" void kernel_launch(void* const* d_in, const int* in_sizes, int n_in,
                              void* d_out, int out_size, void* d_ws, size_t ws_size,
                              hipStream_t stream) {
    (void)in_sizes; (void)n_in; (void)out_size; (void)ws_size;
    const float* x    = (const float*)d_in[0];
    const float* ea   = (const float*)d_in[1];
    const float* neW  = (const float*)d_in[2];  const float* neb  = (const float*)d_in[3];
    const float* eeW  = (const float*)d_in[4];  const float* eeb  = (const float*)d_in[5];
    const float* fW1  = (const float*)d_in[6];  const float* fb1  = (const float*)d_in[7];
    const float* fW2  = (const float*)d_in[8];  const float* fb2  = (const float*)d_in[9];
    const float* gW1  = (const float*)d_in[10]; const float* gb1  = (const float*)d_in[11];
    const float* gW2  = (const float*)d_in[12]; const float* gb2  = (const float*)d_in[13];
    const float* wih  = (const float*)d_in[14]; const float* whh  = (const float*)d_in[15];
    const float* bih  = (const float*)d_in[16]; const float* bhh  = (const float*)d_in[17];
    const float* aW1  = (const float*)d_in[18]; const float* ab1  = (const float*)d_in[19];
    const float* aW2  = (const float*)d_in[20]; const float* ab2  = (const float*)d_in[21];
    const float* npW1 = (const float*)d_in[22]; const float* npb1 = (const float*)d_in[23];
    const float* npW2 = (const float*)d_in[24]; const float* npb2 = (const float*)d_in[25];
    const float* epW1 = (const float*)d_in[26]; const float* epb1 = (const float*)d_in[27];
    const float* epW2 = (const float*)d_in[28]; const float* epb2 = (const float*)d_in[29];
    const int*   ei   = (const int*)d_in[30];   const int*   vt   = (const int*)d_in[31];
    float* out = (float*)d_out;

    char* ws = (char*)d_ws;
    size_t off = 0;
    auto alloc = [&](size_t b) -> char* {
        char* p = ws + off;
        off += (b + 255) & ~(size_t)255;
        return p;
    };
    h16*   x16   = (h16*)  alloc((size_t)NN * NODEF * 2);
    float* hv32  = (float*)alloc((size_t)NN * HH * 4);
    h16*   hv16  = (h16*)  alloc((size_t)NN * HH * 2);
    h16*   he16  = (h16*)  alloc((size_t)EE * HH * 2);
    h16*   cat16 = (h16*)  alloc((size_t)EE * 3 * HH * 2);  // reused as al_in
    h16*   hid16 = (h16*)  alloc((size_t)EE * HH * 2);      // reused for head hiddens
    h16*   m16   = (h16*)  alloc((size_t)EE * HH * 2);
    h16*   a16   = (h16*)  alloc((size_t)EE * HH * 2);
    float* agg   = (float*)alloc((size_t)NN * HH * 4);
    h16*   xcat  = (h16*)  alloc((size_t)NN * 2 * HH * 2);  // reused as np_in
    float* gi    = (float*)alloc((size_t)NN * 3 * HH * 4);
    h16*   wbuf  = (h16*)  alloc((size_t)(2 * HH) * (3 * HH) * 2);
    float* ge    = (float*)alloc(HH * 4);
    float* alph  = (float*)alloc(32 * 4);

    auto nb = [](long n) { return dim3((unsigned)((n + 255) / 256)); };
    auto cvt = [&](const float* i, h16* o, long n) {
        k_f32_to_f16<<<nb(n), 256, 0, stream>>>(i, o, n);
    };
    auto gemm = [&](const h16* A, const h16* B, const float* bias, h16* o16, float* o32,
                    int M, int Nc, int Kc, int relu) {
        dim3 g((Nc + 127) / 128, (M + 127) / 128);
        k_wmma_gemm<<<g, 256, 0, stream>>>(A, B, bias, o16, o32, M, Nc, Kc, relu);
    };

    // node / edge embeddings
    cvt(x, x16, (long)NN * NODEF);
    cvt(neW, wbuf, (long)NODEF * HH);
    gemm(x16, wbuf, neb, hv16, hv32, NN, HH, NODEF, 0);
    k_edge_embed<<<nb((long)EE * HH), 256, 0, stream>>>(ea, eeW, eeb, he16, EE);

    for (int l = 0; l < NLAY; ++l) {
        k_build_cat<<<nb((long)EE * 96), 256, 0, stream>>>(hv16, he16, ei, cat16, EE);
        // f MLP
        cvt(fW1 + (size_t)l * 3 * HH * HH, wbuf, (long)3 * HH * HH);
        gemm(cat16, wbuf, fb1 + (size_t)l * HH, hid16, nullptr, EE, HH, 3 * HH, 1);
        cvt(fW2 + (size_t)l * HH * HH, wbuf, (long)HH * HH);
        gemm(hid16, wbuf, fb2 + (size_t)l * HH, m16, nullptr, EE, HH, HH, 0);
        // g MLP
        cvt(gW1 + (size_t)l * 3 * HH * HH, wbuf, (long)3 * HH * HH);
        gemm(cat16, wbuf, gb1 + (size_t)l * HH, hid16, nullptr, EE, HH, 3 * HH, 1);
        cvt(gW2 + (size_t)l * HH * HH, wbuf, (long)HH * HH);
        gemm(hid16, wbuf, gb2 + (size_t)l * HH, a16, nullptr, EE, HH, HH, 0);
        // aggregate m*a by destination
        k_zero_f32<<<nb((long)NN * HH), 256, 0, stream>>>(agg, (long)NN * HH);
        k_mul_agg<<<nb((long)EE * HH), 256, 0, stream>>>(m16, a16, ei, agg, EE);
        // GRU input gates: gi = [h_v, agg] @ wih^T + bih
        k_build_xcat<<<nb((long)NN * HH), 256, 0, stream>>>(hv32, agg, xcat, NN);
        k_transpose_f32_to_f16<<<nb((long)3 * HH * 2 * HH), 256, 0, stream>>>(
            wih + (size_t)l * 3 * HH * 2 * HH, wbuf, 3 * HH, 2 * HH);
        gemm(xcat, wbuf, bih + (size_t)l * 3 * HH, nullptr, gi, NN, 3 * HH, 2 * HH, 0);
        // sequential scan
        k_gru_scan<<<1, 768, 0, stream>>>(gi, whh + (size_t)l * 3 * HH * HH,
                                          bhh + (size_t)l * 3 * HH, hv32, hv16, NN);
    }

    // graph embedding (column mean)
    k_colmean<<<HH, 256, 0, stream>>>(hv32, ge, NN);

    // node prediction head -> p_v
    k_build_npin<<<nb((long)NN * HH), 256, 0, stream>>>(ge, hv32, xcat, NN);
    cvt(npW1, wbuf, (long)2 * HH * HH);
    gemm(xcat, wbuf, npb1, hid16, nullptr, NN, HH, 2 * HH, 1);
    k_node_pred<<<nb(NN), 256, 0, stream>>>(hid16, npW2, npb2, out, NN);

    // alpha head
    k_build_alin<<<nb((long)NN * HH), 256, 0, stream>>>(ge, hv32, vt, cat16, NN);
    cvt(aW1, wbuf, (long)3 * HH * HH);
    gemm(cat16, wbuf, ab1, hid16, nullptr, NN, HH, 3 * HH, 1);
    k_zero_f32<<<1, 32, 0, stream>>>(alph, KKm);
    k_alpha_acc<<<nb(NN), 256, 0, stream>>>(hid16, aW2, ab2, alph, NN);
    k_alpha_softmax<<<1, 32, 0, stream>>>(alph);

    // edge prediction head -> p_e
    cvt(epW1, wbuf, (long)HH * HH);
    gemm(hv16, wbuf, epb1, hid16, nullptr, NN, HH, HH, 1);
    k_zero_f32<<<nb((long)NN * NETT), 256, 0, stream>>>(out + (size_t)NN * NNTT, (long)NN * NETT);
    k_edge_pred<<<nb((long)NN * KKm), 256, 0, stream>>>(hid16, epW2, epb2, alph,
                                                        out + (size_t)NN * NNTT, NN);
}
